// PointerAttention_12799002542779
// MI455X (gfx1250) — matrix operand order repaired and checked by reading, not compiled, not run on hardware
//
#include <hip/hip_runtime.h>
#include <hip/hip_bf16.h>
#include <math.h>

#define Bb 32
#define Tt 2048
#define Dd 512
#define Uu 512
#define TM 128             // value rows per workgroup (8 waves x 16 rows)
#define LPAD 8
#define LROW (Dd + LPAD)   // 520 halfs per padded LDS row
#define NCH (Uu / 16)      // 32 N-chunks

typedef __attribute__((ext_vector_type(16))) __bf16 v16bf;
typedef __attribute__((ext_vector_type(8)))  __bf16 v8bf;
typedef __attribute__((ext_vector_type(8)))  float  v8f;

static __device__ __forceinline__ unsigned short f2bf(float f) {
  unsigned int x = __float_as_uint(f);
  x += 0x7FFFu + ((x >> 16) & 1u);          // round-to-nearest-even
  return (unsigned short)(x >> 16);
}

// ---------------- Kernel 0: W1 [D][U] f32 -> w1t [U][D] bf16 ----------------
__global__ __launch_bounds__(256) void conv_w1_kernel(
    const float* __restrict__ W1, unsigned short* __restrict__ w1t) {
  int i = blockIdx.x * 256 + threadIdx.x;   // i = u*512 + d
  int u = i >> 9;
  int d = i & 511;
  w1t[i] = f2bf(W1[d * Uu + u]);
}

// ---------------- Kernel 1: w2q[b][u] = query[b] @ W2 ----------------
__global__ __launch_bounds__(256) void w2q_kernel(
    const float* __restrict__ query, const float* __restrict__ W2,
    float* __restrict__ w2q) {
  const int b = blockIdx.x;
  const int tid = threadIdx.x;
  __shared__ float q[Dd];
  for (int d = tid; d < Dd; d += 256) q[d] = query[b * Dd + d];
  __syncthreads();
  for (int u = tid; u < Uu; u += 256) {
    float acc = 0.f;
    for (int d = 0; d < Dd; ++d) acc += q[d] * W2[d * Uu + u];
    w2q[b * Uu + u] = acc;
  }
}

// ---------------- Kernel 2: fused GEMM + tanh + V-dot -> u[B,T] ----------------
__global__ __launch_bounds__(256) void attn_gemm_kernel(
    const float* __restrict__ value, const unsigned short* __restrict__ w1t,
    const float* __restrict__ w2q, const float* __restrict__ Vv,
    float* __restrict__ u_out) {
  __shared__ unsigned short As[TM][LROW];     // 128x520 halfs = 133,120 B
  __shared__ unsigned short Bs[2][16][LROW];  // 2x16x520 halfs = 33,280 B

  const int b   = blockIdx.y;
  const int t0  = blockIdx.x * TM;
  const int tid  = threadIdx.x;
  const int lane = tid & 31;
  const int wave = tid >> 5;

  const unsigned long long w1t64 = (unsigned long long)w1t;
  const unsigned bsBase[2] = {
      (unsigned)(size_t)&Bs[0][0][0],   // flat low 32 bits == LDS byte offset
      (unsigned)(size_t)&Bs[1][0][0]};

  // Async-stage a 16-col W1 slab (512 K x 16 N, bf16) into Bs[buf].
  // 1024 b128 lane-transfers / 32 lanes / 8 waves = 4 async ops per wave.
  auto issue_chunk = [&](int nc, int buf) {
    const int n0 = nc * 16;
    const unsigned base = bsBase[buf];
    #pragma unroll
    for (int it = 0; it < 4; ++it) {
      const int i = tid + it * 256;   // 0..1023
      const int c = i >> 6;           // col 0..15
      const int q = i & 63;           // uint4 within col
      const unsigned ldsoff = base + (unsigned)((c * LROW + q * 8) * 2);
      const unsigned goff   = (unsigned)(((n0 + c) * Dd + q * 8) * 2);
      asm volatile("global_load_async_to_lds_b128 %0, %1, %2"
                   :: "v"(ldsoff), "v"(goff), "s"(w1t64) : "memory");
    }
  };

  // Kick off chunk 0 before A staging so it overlaps the 256 KB value read.
  issue_chunk(0, 0);

  // ---- Stage A tile: value[b, t0:t0+128, :] f32 -> bf16 in LDS (once) ----
  const float4* src = reinterpret_cast<const float4*>(
      value + ((size_t)b * Tt + t0) * Dd);
  #pragma unroll 4
  for (int i = tid; i < TM * Dd / 4; i += 256) {
    int r = i >> 7;          // 128 float4 per row
    int q = i & 127;
    float4 v = src[i];
    ushort4 h;
    h.x = f2bf(v.x); h.y = f2bf(v.y); h.z = f2bf(v.z); h.w = f2bf(v.w);
    *reinterpret_cast<ushort4*>(&As[r][q * 4]) = h;
  }

  // Per-lane fragment geometry (ISA 7.12.2, wave32)
  const int rowA  = wave * 16 + (lane & 15);
  const int kofsA = (lane & 16) ? 8 : 0;          // A K-pair lane-half offset
  const int colB  = lane & 15;                    // B: N = lane & 15
  const int kofsB = (lane & 16) ? 8 : 0;          // B K lane-half offset
  const unsigned short* Arow = &As[rowA][0];

  float upart[8] = {0.f, 0.f, 0.f, 0.f, 0.f, 0.f, 0.f, 0.f};

  for (int nc = 0; nc < NCH; ++nc) {
    const int buf = nc & 1;
    // All waves finished reading Bs[buf^1] (previous compute) -> safe to refill.
    __syncthreads();
    if (nc + 1 < NCH) {
      issue_chunk(nc + 1, buf ^ 1);
      asm volatile("s_wait_asynccnt 0x4" ::: "memory");  // chunk nc landed (in-order)
    } else {
      asm volatile("s_wait_asynccnt 0x0" ::: "memory");
    }
    __syncthreads();  // every wave's async-loaded portion of chunk nc visible

    const unsigned short* Bcol = &Bs[buf][colB][0];
    v8f acc = {0.f, 0.f, 0.f, 0.f, 0.f, 0.f, 0.f, 0.f};
    #pragma unroll
    for (int ks = 0; ks < Dd / 32; ++ks) {
      const int kb = ks * 32;
      v8bf a0 = *reinterpret_cast<const v8bf*>(Arow + kb + kofsA);
      v8bf a1 = *reinterpret_cast<const v8bf*>(Arow + kb + 16 + kofsA);
      v8bf b0 = *reinterpret_cast<const v8bf*>(Bcol + kb + kofsB);
      v8bf b1 = *reinterpret_cast<const v8bf*>(Bcol + kb + 16 + kofsB);
      v16bf a = __builtin_shufflevector(a0, a1, 0,1,2,3,4,5,6,7,8,9,10,11,12,13,14,15);
      v16bf bb = __builtin_shufflevector(b0, b1, 0,1,2,3,4,5,6,7,8,9,10,11,12,13,14,15);
      acc = __builtin_amdgcn_wmma_f32_16x16x32_bf16(
          false, a, false, bb, (short)0, acc, false, false);
    }

    // Fused epilogue for this 16-wide N chunk: V[n] * tanh(c + w2q[n])
    const int n   = nc * 16 + colB;
    const float w2 = w2q[b * Uu + n];
    const float vv = Vv[n];
    #pragma unroll
    for (int i = 0; i < 8; ++i)
      upart[i] += vv * tanhf(acc[i] + w2);
  }

  // Reduce across the 16 lanes (columns) within each lane-half
  #pragma unroll
  for (int m = 1; m <= 8; m <<= 1) {
    #pragma unroll
    for (int i = 0; i < 8; ++i)
      upart[i] += __shfl_xor(upart[i], m, 32);
  }
  if ((lane & 15) == 0) {
    // C layout: VGPR i holds rows i (lanes 0-15) and i+8 (lanes 16-31)
    const int rbase = t0 + wave * 16 + ((lane & 16) ? 8 : 0);
    #pragma unroll
    for (int i = 0; i < 8; ++i)
      u_out[(size_t)b * Tt + rbase + i] = upart[i];
  }
}

// ---------------- Kernel 3: masked softmax + argmax + gather ----------------
__global__ __launch_bounds__(256) void finalize_kernel(
    const float* __restrict__ value, const int* __restrict__ mask,
    float* __restrict__ a_io /* in: u, out: a */, float* __restrict__ ctx) {
  const int b = blockIdx.x;
  const int tid = threadIdx.x;
  __shared__ float redf[256];
  __shared__ int   redi[256];

  float s[8];
  float m = -3.0e38f;
  #pragma unroll
  for (int j = 0; j < 8; ++j) {
    const int t = tid + j * 256;
    float v = a_io[(size_t)b * Tt + t];
    if (mask[b * Tt + t] == 0) v -= 1.0e20f;
    s[j] = v;
    m = fmaxf(m, v);
  }
  redf[tid] = m;
  __syncthreads();
  for (int off = 128; off > 0; off >>= 1) {
    if (tid < off) redf[tid] = fmaxf(redf[tid], redf[tid + off]);
    __syncthreads();
  }
  const float M = redf[0];
  __syncthreads();

  float e[8];
  float sum = 0.f;
  #pragma unroll
  for (int j = 0; j < 8; ++j) { e[j] = __expf(s[j] - M); sum += e[j]; }
  redf[tid] = sum;
  __syncthreads();
  for (int off = 128; off > 0; off >>= 1) {
    if (tid < off) redf[tid] += redf[tid + off];
    __syncthreads();
  }
  const float inv = 1.0f / redf[0];
  __syncthreads();
  #pragma unroll
  for (int j = 0; j < 8; ++j)
    a_io[(size_t)b * Tt + tid + j * 256] = e[j] * inv;

  // First-index argmax over masked scores (softmax is monotone in s)
  float bv = -3.0e38f;
  int bi = Tt;
  #pragma unroll
  for (int j = 0; j < 8; ++j) {
    const int t = tid + j * 256;
    if (s[j] > bv) { bv = s[j]; bi = t; }
  }
  redf[tid] = bv;
  redi[tid] = bi;
  __syncthreads();
  for (int off = 128; off > 0; off >>= 1) {
    if (tid < off) {
      float ov = redf[tid + off];
      int   oi = redi[tid + off];
      if (ov > redf[tid] || (ov == redf[tid] && oi < redi[tid])) {
        redf[tid] = ov; redi[tid] = oi;
      }
    }
    __syncthreads();
  }
  const int idx = redi[0];
  for (int d = tid; d < Dd; d += 256)
    ctx[b * Dd + d] = value[((size_t)b * Tt + idx) * Dd + d];
}

extern "C" void kernel_launch(void* const* d_in, const int* in_sizes, int n_in,
                              void* d_out, int out_size, void* d_ws, size_t ws_size,
                              hipStream_t stream) {
  const float* value = (const float*)d_in[0];
  const float* query = (const float*)d_in[1];
  const int*   mask  = (const int*)d_in[2];
  const float* W1    = (const float*)d_in[3];
  const float* W2    = (const float*)d_in[4];
  const float* Vv    = (const float*)d_in[5];

  // Workspace: w1t bf16 [U][D] (512 KB) + w2q f32 [B][U] (64 KB)
  unsigned short* w1t = (unsigned short*)d_ws;
  float* w2q = (float*)((char*)d_ws + (size_t)Uu * Dd * sizeof(unsigned short));

  float* ctx = (float*)d_out;            // [B, D]
  float* a   = (float*)d_out + Bb * Dd;  // [B, T] — stages u, then holds a

  conv_w1_kernel<<<(Uu * Dd) / 256, 256, 0, stream>>>(W1, w1t);
  w2q_kernel<<<Bb, 256, 0, stream>>>(query, W2, w2q);
  dim3 g2(Tt / TM, Bb);
  attn_gemm_kernel<<<g2, 256, 0, stream>>>(value, w1t, w2q, Vv, a);
  finalize_kernel<<<Bb, 256, 0, stream>>>(value, mask, a, ctx);
}
